// GAT_20564303414088
// MI455X (gfx1250) — compile-verified, matrix operand order checked by reading, not run on hardware
//
#include <hip/hip_runtime.h>
#include <hip/hip_bf16.h>
#include <math.h>

#define HEADS 4
#define CH 64
#define FDIM 256            // HEADS*CH
#define NEG_SLOPE 0.2f

typedef __attribute__((ext_vector_type(16))) _Float16 v16h;
typedef __attribute__((ext_vector_type(8)))  _Float16 v8h;
typedef __attribute__((ext_vector_type(8)))  float    v8f;

// order-preserving float<->uint encoding so integer atomicMax gives float max
__device__ __forceinline__ unsigned enc_f32(float v) {
    unsigned b = __float_as_uint(v);
    return (b & 0x80000000u) ? ~b : (b | 0x80000000u);
}
__device__ __forceinline__ float dec_f32(unsigned k) {
    return __uint_as_float((k & 0x80000000u) ? (k ^ 0x80000000u) : ~k);
}
#define ENC_NEG_INF 0x007FFFFFu   // enc(-inf)

// ---------------- conversion kernels ----------------
__global__ void k_cvt_f16(const float* __restrict__ x, _Float16* __restrict__ xh, int total) {
    int i = blockIdx.x * blockDim.x + threadIdx.x;
    if (i < total) xh[i] = (_Float16)x[i];
}

// W[k][n] (row-major 256x256) -> Wth[n][k] so B tiles load contiguously
__global__ void k_cvt_wt(const float* __restrict__ W, _Float16* __restrict__ Wth) {
    int k = blockIdx.x, n = threadIdx.x;
    Wth[n * FDIM + k] = (_Float16)W[k * FDIM + n];
}

// ---------------- WMMA GEMM: H[N,256] = Xh[N,256] @ W[256,256] ----------------
// block = 256 threads (8 waves). grid = ((N+15)/16, 2). wave w of block (x,y)
// computes the 16x16 output tile (rowTile = blockIdx.x, colTile = y*8 + w).
__global__ __launch_bounds__(256) void k_gemm_wmma(
    const _Float16* __restrict__ Ah, const _Float16* __restrict__ Bth,
    float* __restrict__ Hout, int N)
{
    int wave = threadIdx.x >> 5;
    int lane = threadIdx.x & 31;
    int m    = lane & 15;      // row (A) / col (B) owned by this lane
    int half = lane >> 4;      // K sub-block selector per ISA 16-bit layout
    int row  = blockIdx.x * 16 + m;
    int rclp = row < N ? row : N - 1;
    int col0 = (blockIdx.y * 8 + wave) * 16;

    const _Float16* aRow = Ah  + (size_t)rclp * FDIM;
    const _Float16* bRow = Bth + (size_t)(col0 + m) * FDIM;

    v8f acc = {};
#pragma unroll
    for (int k0 = 0; k0 < FDIM; k0 += 32) {
        // A 16x32 f16 layout: lane(m,half) holds K = k0+half*8+[0..7] and k0+16+half*8+[0..7]
        v8h alo = *(const v8h*)(aRow + k0 + half * 8);
        v8h ahi = *(const v8h*)(aRow + k0 + 16 + half * 8);
        v8h blo = *(const v8h*)(bRow + k0 + half * 8);
        v8h bhi = *(const v8h*)(bRow + k0 + 16 + half * 8);
        v16h a, b;
#pragma unroll
        for (int i = 0; i < 8; ++i) {
            a[i] = alo[i]; a[8 + i] = ahi[i];
            b[i] = blo[i]; b[8 + i] = bhi[i];
        }
        acc = __builtin_amdgcn_wmma_f32_16x16x32_f16(
            /*neg_a=*/false, a, /*neg_b=*/false, b,
            /*c_mod=*/(short)0, acc, /*reuse_a=*/false, /*reuse_b=*/false);
    }
    // D layout: lane n = lane&15; VGPR r -> output row r + 8*half
#pragma unroll
    for (int r = 0; r < 8; ++r) {
        int orow = blockIdx.x * 16 + r + 8 * half;
        if (orow < N) Hout[(size_t)orow * FDIM + col0 + m] = acc[r];
    }
}

// ---------------- attention scores: s = <h[n,h,:], a[h,:]> ----------------
__global__ void k_scores(const float* __restrict__ h,
                         const float* __restrict__ a_s, const float* __restrict__ a_d,
                         float* __restrict__ ssrc, float* __restrict__ sdst, int N)
{
    int idx = blockIdx.x * blockDim.x + threadIdx.x;  // node*HEADS + head
    if (idx >= N * HEADS) return;
    int node = idx >> 2, hd = idx & 3;
    const float* hp = h + (size_t)node * FDIM + hd * CH;
    const float* as = a_s + hd * CH;
    const float* ad = a_d + hd * CH;
    float s1 = 0.f, s2 = 0.f;
#pragma unroll 4
    for (int c = 0; c < CH; ++c) { float v = hp[c]; s1 += v * as[c]; s2 += v * ad[c]; }
    ssrc[idx] = s1;
    sdst[idx] = s2;
}

// ---------------- per-layer state init ----------------
__global__ void k_init(float* __restrict__ agg, unsigned* __restrict__ menc,
                       float* __restrict__ den, int N)
{
    int i = blockIdx.x * blockDim.x + threadIdx.x;
    if (i < N * FDIM) agg[i] = 0.f;
    if (i < N * HEADS) { menc[i] = ENC_NEG_INF; den[i] = 0.f; }
}

// ---------------- edge pass 1: leaky-relu logits + segment max ----------------
__global__ void k_edge_max(const int* __restrict__ src, const int* __restrict__ dst,
                           const float* __restrict__ ssrc, const float* __restrict__ sdst,
                           float* __restrict__ elog, unsigned* __restrict__ menc,
                           int E, int NE)
{
    int idx = blockIdx.x * blockDim.x + threadIdx.x;  // edge*HEADS + head
    if (idx >= NE * HEADS) return;
    int e = idx >> 2, hd = idx & 3;
    int s, d;
    if (e < E) { s = src[e]; d = dst[e]; } else { s = d = e - E; }  // self-loops
    float v = ssrc[s * HEADS + hd] + sdst[d * HEADS + hd];
    v = (v > 0.f) ? v : NEG_SLOPE * v;
    elog[idx] = v;
    atomicMax(&menc[d * HEADS + hd], enc_f32(v));
}

// ---------------- edge pass 2: exp(e - max) + segment sum ----------------
__global__ void k_edge_exp(const int* __restrict__ src, const int* __restrict__ dst,
                           float* __restrict__ elog, const unsigned* __restrict__ menc,
                           float* __restrict__ den, int E, int NE)
{
    int idx = blockIdx.x * blockDim.x + threadIdx.x;
    if (idx >= NE * HEADS) return;
    int e = idx >> 2, hd = idx & 3;
    int d = (e < E) ? dst[e] : e - E;
    float mm = dec_f32(menc[d * HEADS + hd]);
    float v = __expf(elog[idx] - mm);
    elog[idx] = v;
    atomicAdd(&den[d * HEADS + hd], v);
}

// ---------------- edge pass 3: scatter alpha * h[src] into agg[dst] ----------------
__global__ __launch_bounds__(256) void k_scatter(
    const int* __restrict__ src, const int* __restrict__ dst,
    const float* __restrict__ elog, const float* __restrict__ den,
    const float* __restrict__ h, float* __restrict__ agg, int E)
{
    int e = blockIdx.x;              // one block per edge
    int t = threadIdx.x;             // channel 0..255
    int hd = t >> 6;
    int s, d;
    if (e < E) { s = src[e]; d = dst[e]; } else { s = d = e - E; }
    float alpha = elog[e * HEADS + hd] / den[d * HEADS + hd];
    float val = h[(size_t)s * FDIM + t] * alpha;
    atomicAdd(&agg[(size_t)d * FDIM + t], val);
}

// ---------------- finalize: +bias then ReLU (layers 1,2) ----------------
__global__ void k_fin_relu(const float* __restrict__ agg, const float* __restrict__ bias,
                           float* __restrict__ out, int total)
{
    int i = blockIdx.x * blockDim.x + threadIdx.x;
    if (i >= total) return;
    out[i] = fmaxf(agg[i] + bias[i & (FDIM - 1)], 0.f);
}

// ---------------- finalize: +bias then L2 normalize (layer 3) ----------------
__global__ __launch_bounds__(256) void k_fin_norm(
    const float* __restrict__ agg, const float* __restrict__ bias, float* __restrict__ out)
{
    __shared__ float red[8];
    int node = blockIdx.x, t = threadIdx.x;
    float v = agg[(size_t)node * FDIM + t] + bias[t];
    float ss = v * v;
#pragma unroll
    for (int o = 16; o > 0; o >>= 1) ss += __shfl_down(ss, o);   // wave32 reduce
    if ((t & 31) == 0) red[t >> 5] = ss;
    __syncthreads();
    if (t == 0) {
        float s = 0.f;
#pragma unroll
        for (int i = 0; i < 8; ++i) s += red[i];
        red[0] = s;
    }
    __syncthreads();
    float nrm = fmaxf(sqrtf(red[0]), 1e-12f);
    out[(size_t)node * FDIM + t] = v / nrm;
}

extern "C" void kernel_launch(void* const* d_in, const int* in_sizes, int n_in,
                              void* d_out, int out_size, void* d_ws, size_t ws_size,
                              hipStream_t stream)
{
    (void)n_in; (void)out_size; (void)ws_size;
    const int N  = in_sizes[0] / FDIM;
    const int E  = in_sizes[1] / 2;
    const int NE = E + N;                      // + self loops

    const float* x0  = (const float*)d_in[0];
    const int*   src = (const int*)d_in[1];
    const int*   dst = src + E;

    char* ws = (char*)d_ws;
    size_t off = 0;
    auto alloc = [&](size_t bytes) -> char* {
        char* p = ws + off;
        off += (bytes + 255) & ~(size_t)255;
        return p;
    };
    _Float16* xh   = (_Float16*)alloc((size_t)N * FDIM * 2);
    _Float16* Wth  = (_Float16*)alloc((size_t)FDIM * FDIM * 2);
    float*    h    = (float*)   alloc((size_t)N * FDIM * 4);
    float*    ssrc = (float*)   alloc((size_t)N * HEADS * 4);
    float*    sdst = (float*)   alloc((size_t)N * HEADS * 4);
    unsigned* menc = (unsigned*)alloc((size_t)N * HEADS * 4);
    float*    den  = (float*)   alloc((size_t)N * HEADS * 4);
    float*    elog = (float*)   alloc((size_t)NE * HEADS * 4);
    float*    agg  = (float*)   alloc((size_t)N * FDIM * 4);
    float*    xA   = (float*)   alloc((size_t)N * FDIM * 4);

    const int T = 256;
    const int gN256 = (N * FDIM + T - 1) / T;
    const int gNH   = (N * HEADS + T - 1) / T;
    const int gEH   = (NE * HEADS + T - 1) / T;

    for (int L = 0; L < 3; ++L) {
        const float* xin = (L == 0) ? x0 : xA;
        const float* W   = (const float*)d_in[2 + L * 4];
        const float* a_s = (const float*)d_in[3 + L * 4];
        const float* a_d = (const float*)d_in[4 + L * 4];
        const float* bia = (const float*)d_in[5 + L * 4];
        float* outp = (L == 2) ? (float*)d_out : xA;

        k_cvt_f16<<<gN256, T, 0, stream>>>(xin, xh, N * FDIM);
        k_cvt_wt<<<FDIM, FDIM, 0, stream>>>(W, Wth);
        k_gemm_wmma<<<dim3((N + 15) / 16, 2), T, 0, stream>>>(xh, Wth, h, N);
        k_scores<<<gNH, T, 0, stream>>>(h, a_s, a_d, ssrc, sdst, N);
        k_init<<<gN256, T, 0, stream>>>(agg, menc, den, N);
        k_edge_max<<<gEH, T, 0, stream>>>(src, dst, ssrc, sdst, elog, menc, E, NE);
        k_edge_exp<<<gEH, T, 0, stream>>>(src, dst, elog, menc, den, E, NE);
        k_scatter<<<NE, T, 0, stream>>>(src, dst, elog, den, h, agg, E);
        if (L < 2) k_fin_relu<<<gN256, T, 0, stream>>>(agg, bia, outp, N * FDIM);
        else       k_fin_norm<<<N, T, 0, stream>>>(agg, bia, outp);
    }
}
